// DPModel_79757542687218
// MI455X (gfx1250) — matrix operand order repaired and checked by reading, not compiled, not run on hardware
//
#include <hip/hip_runtime.h>
#include <hip/hip_bf16.h>
#include <math.h>

typedef __attribute__((ext_vector_type(16))) _Float16 v16h;
typedef __attribute__((ext_vector_type(8)))  float    v8f;

#define WMMA_F16(a, b, c) \
  __builtin_amdgcn_wmma_f32_16x16x32_f16(false, (a), false, (b), (short)0, (c), false, false)

// Branchless tanh: gfx1250 V_TANH_F32 (confirmed lowering), else exp2-based
__device__ __forceinline__ float fast_tanh(float x) {
#if __has_builtin(__builtin_amdgcn_tanhf)
  return __builtin_amdgcn_tanhf(x);
#else
  float xc = fminf(fmaxf(x, -16.0f), 16.0f);
  float a  = __builtin_amdgcn_exp2f(xc * 2.885390081777927f);   // e^(2x)
  return (a - 1.0f) * __builtin_amdgcn_rcpf(a + 1.0f);
#endif
}

// ---- CDNA5 async global->LDS copy (ASYNCcnt path), 16B granularity ----
__device__ __forceinline__ void async_ld_b128(void* lds_dst, const void* gsrc) {
  auto lp = (__attribute__((address_space(3))) char*)lds_dst;   // 32-bit LDS offset
  asm volatile("global_load_async_to_lds_b128 %0, %1, off"
               :: "v"(lp), "v"(gsrc) : "memory");
}
__device__ __forceinline__ void wait_async0() {
  asm volatile("s_wait_asynccnt 0" ::: "memory");
}

// ---- WMMA fragment loaders (contiguous k order, same for A and B) ----
// A 16x32 (MxK) row-major: lane L -> row M=L%16; elem i -> K=i+16*(L/16)
__device__ __forceinline__ v16h frag_A(const _Float16* __restrict__ p, int row0, int ld, int lane) {
  int r  = row0 + (lane & 15);
  int hl = lane >> 4;
  v16h a;
#pragma unroll
  for (int i = 0; i < 16; ++i) {
    int k = i + (hl << 4);
    a[i] = p[r * ld + k];
  }
  return a;
}
// B 32x16 (KxN) stored TRANSPOSED (N-major): lane L -> col N=L%16, elem i -> K=i+16*(L/16)
__device__ __forceinline__ v16h frag_BT(const _Float16* __restrict__ p, int col0, int ld, int lane) {
  int n  = col0 + (lane & 15);
  int k0 = (lane >> 4) << 4;
  v16h b;
#pragma unroll
  for (int i = 0; i < 16; ++i) b[i] = p[n * ld + k0 + i];
  return b;
}
// C/D f32 16x16: VGPR v -> row = v + 8*(lane/16), col = lane%16

// =====================================================================
// Kernel 1: fused geometry + embedding MLP + T = RX^T G  (per atom)
// =====================================================================
__global__ __launch_bounds__(256) void k_embed(
    const float* __restrict__ coord, const int* __restrict__ nbrs,
    const float* __restrict__ We1, const float* __restrict__ be1,
    const float* __restrict__ We2, const float* __restrict__ be2,
    const float* __restrict__ We3, const float* __restrict__ be3,
    float* __restrict__ Tout)
{
  __shared__ float    s_lds[128];
  __shared__ _Float16 RXl[16 * 128];         // rows 4..15 zero-padded
  __shared__ _Float16 h2s[128 * 64];         // A operand, row-major
  __shared__ _Float16 W3T[128 * 64];         // We3 transposed [n][k]
  __shared__ _Float16 scratch[8192];         // phase B: h1 + We2T; phase C: 8 x GcolT(16x64)
  __shared__ float    be2s[64], be3s[128];

  const int n = blockIdx.x;
  const int t = threadIdx.x;
  const int lane = t & 31, w = t >> 5, hl = lane >> 4;

  const float cx = coord[3 * n + 0], cy = coord[3 * n + 1], cz = coord[3 * n + 2];
  if (t < 128) {
    int j = nbrs[n * 128 + t];
    float x = coord[3 * j + 0] - cx, y = coord[3 * j + 1] - cy, z = coord[3 * j + 2] - cz;
    float r = __builtin_amdgcn_sqrtf(x * x + y * y + z * z);
    float u = (r - 0.5f) * (1.0f / 5.5f);
    u = fminf(fmaxf(u, 0.0f), 1.0f);
    float sw  = 0.5f * __builtin_amdgcn_cosf(0.5f * u) + 0.5f;   // cos(pi*u)
    float inv = __builtin_amdgcn_rcpf(r + 1e-16f);
    float sr  = sw * inv;
    float srn = sr * 10.0f;                      // / SR_STD
    s_lds[t]  = (sr - 0.05f) * 10.0f;            // sr_center
    const float q = 1.7320508075688772f;         // sqrt(3)
    RXl[0 * 128 + t] = (_Float16)srn;
    RXl[1 * 128 + t] = (_Float16)(q * srn * x * inv);
    RXl[2 * 128 + t] = (_Float16)(q * srn * y * inv);
    RXl[3 * 128 + t] = (_Float16)(q * srn * z * inv);
  }
  for (int i = t; i < 12 * 128; i += 256) RXl[512 + i] = (_Float16)0.0f;
  for (int i = t; i < 64 * 32; i += 256) {                // We2T [n][k]
    int nn = i >> 5, k = i & 31;
    scratch[4096 + i] = (_Float16)We2[k * 64 + nn];
  }
  for (int i = t; i < 128 * 64; i += 256) {               // We3T [n][k]
    int nn = i >> 6, k = i & 63;
    W3T[i] = (_Float16)We3[k * 128 + nn];
  }
  if (t < 64)  be2s[t] = be2[t];
  if (t < 128) be3s[t] = be3[t];
  __syncthreads();

  // h1 = tanh(s * We1 + be1): 128 rows x 32 cols
  {
    int m = t >> 1, c0 = (t & 1) * 16;
    float s = s_lds[m];
#pragma unroll
    for (int c = 0; c < 16; ++c) {
      int col = c0 + c;
      scratch[m * 32 + col] = (_Float16)fast_tanh(s * We1[col] + be1[col]);
    }
  }
  __syncthreads();

  // GEMM2: h2 = dup(h1) + tanh(h1 @ We2 + be2); wave w owns rows [16w,16w+16)
  {
    const _Float16* h1  = scratch;
    const _Float16* W2T = scratch + 4096;
    v16h a = frag_A(h1, 16 * w, 32, lane);
#pragma unroll
    for (int nt = 0; nt < 4; ++nt) {
      v16h b = frag_BT(W2T, 16 * nt, 32, lane);
      v8f c = {};
      c = WMMA_F16(a, b, c);
      int col = 16 * nt + (lane & 15);
#pragma unroll
      for (int v = 0; v < 8; ++v) {
        int r = 16 * w + v + 8 * hl;
        float val = (float)h1[r * 32 + (col & 31)] + fast_tanh(c[v] + be2s[col]);
        h2s[r * 64 + col] = (_Float16)val;
      }
    }
  }
  __syncthreads();

  // GEMM3 fused with T-einsum: wave w owns G columns [16w,16w+16)
  {
    _Float16* GcolT = scratch + w * 1024;   // 16 cols x 64 rows (transposed slab)
    const int cl = lane & 15;
    v8f tacc = {};
    for (int hb = 0; hb < 2; ++hb) {
#pragma unroll
      for (int mt = 0; mt < 4; ++mt) {
        int row0 = 64 * hb + 16 * mt;
        v8f c = {};
#pragma unroll
        for (int ks = 0; ks < 2; ++ks) {
          v16h a = frag_A(h2s + 32 * ks, row0, 64, lane);
          v16h b = frag_BT(W3T + 32 * ks, 16 * w, 64, lane);
          c = WMMA_F16(a, b, c);
        }
        int col = 16 * w + cl;
#pragma unroll
        for (int v = 0; v < 8; ++v) {
          int r = row0 + v + 8 * hl;
          float g = (float)h2s[r * 64 + (col & 63)] + fast_tanh(c[v] + be3s[col]);
          GcolT[cl * 64 + (r - 64 * hb)] = (_Float16)g;   // same-wave LDS: in-order
        }
      }
#pragma unroll
      for (int ks = 0; ks < 2; ++ks) {
        int kb = 64 * hb + 32 * ks;
        v16h a = frag_A(RXl + kb, 0, 128, lane);
        v16h b = frag_BT(GcolT + 32 * ks, 0, 64, lane);
        tacc = WMMA_F16(a, b, tacc);
      }
    }
    if (hl == 0) {
#pragma unroll
      for (int v = 0; v < 4; ++v)
        Tout[n * 512 + v * 128 + 16 * w + cl] = tacc[v];
    }
  }
}

// =====================================================================
// Kernel 2: D[n][a*128+w] = T0[w]T0[a] + sum_k T3[k][w]T3[k][a]  (f16)
// =====================================================================
__global__ __launch_bounds__(256) void k_descriptor(
    const float* __restrict__ Tin, const float* __restrict__ Tbias,
    _Float16* __restrict__ D)
{
  __shared__ float T0[128], T3[3 * 128];
  int n = blockIdx.x, t = threadIdx.x;
  const float inv = 1.0f / 128.0f;
  for (int i = t; i < 512; i += 256) {
    float v = Tin[n * 512 + i] * inv;
    if (i < 128) T0[i] = v + Tbias[i];
    else         T3[i - 128] = v;
  }
  __syncthreads();
#pragma unroll
  for (int q = 0; q < 8; ++q) {
    int idx = t * 8 + q;
    int a = idx >> 7, wc = idx & 127;
    float val = T0[wc] * T0[a]
              + T3[0 * 128 + wc] * T3[0 * 128 + a]
              + T3[1 * 128 + wc] * T3[1 * 128 + a]
              + T3[2 * 128 + wc] * T3[2 * 128 + a];
    D[(size_t)n * 2048 + idx] = (_Float16)val;
  }
}

// =====================================================================
// Weight prep: transpose fit weights to f16, B-fragment friendly [n][k]
// WT1: 240 x 2048 ; WT2/WT3: 240 x 256 (k 240..255 zero)
// =====================================================================
__global__ __launch_bounds__(256) void k_prep_w(
    const float* __restrict__ Wf1, const float* __restrict__ Wf2,
    const float* __restrict__ Wf3,
    _Float16* __restrict__ WT1, _Float16* __restrict__ WT2,
    _Float16* __restrict__ WT3)
{
  int tid = blockIdx.x * 256 + threadIdx.x;
  int stride = gridDim.x * 256;
  for (int i = tid; i < 240 * 2048; i += stride) {
    int nn = i >> 11, k = i & 2047;
    WT1[i] = (_Float16)Wf1[(size_t)k * 240 + nn];
  }
  for (int i = tid; i < 240 * 256; i += stride) {
    int nn = i >> 8, k = i & 255;
    _Float16 v2 = (k < 240) ? (_Float16)Wf2[k * 240 + nn] : (_Float16)0.0f;
    _Float16 v3 = (k < 240) ? (_Float16)Wf3[k * 240 + nn] : (_Float16)0.0f;
    WT2[i] = v2;
    WT3[i] = v3;
  }
}

// =====================================================================
// Kernel 3: H = tanh(D @ Wf1 + bf1); M=4096, K=2048, N=240 (pad to 256)
// A tiles (D) and B tiles (WT1) staged with async global->LDS copies
// =====================================================================
__global__ __launch_bounds__(256) void k_fit1(
    const _Float16* __restrict__ D, const _Float16* __restrict__ WT1,
    const float* __restrict__ bf1, _Float16* __restrict__ H)
{
  __shared__ _Float16 Ash[128 * 32];
  __shared__ _Float16 BshT[240 * 32];
  int t = threadIdx.x, lane = t & 31, w = t >> 5, hl = lane >> 4;
  int rowBase = blockIdx.x * 128;
  v8f zero = {};
  v8f acc[15];
#pragma unroll
  for (int i = 0; i < 15; ++i) acc[i] = zero;

  for (int kk = 0; kk < 64; ++kk) {
    {   // A: 128 rows x 32 halves (two b128 per thread)
      int row = t >> 1, seg = (t & 1) * 16;
      const _Float16* src = D + (size_t)(rowBase + row) * 2048 + kk * 32 + seg;
      async_ld_b128(&Ash[row * 32 + seg],     src);
      async_ld_b128(&Ash[row * 32 + seg + 8], src + 8);
    }
    // B: 240 rows x 32 halves = 960 b128 chunks
    for (int c = t; c < 960; c += 256) {
      int row = c >> 2, seg = (c & 3) * 8;
      async_ld_b128(&BshT[row * 32 + seg], WT1 + (size_t)row * 2048 + kk * 32 + seg);
    }
    wait_async0();
    __syncthreads();
    v16h a = frag_A(Ash, 16 * w, 32, lane);
#pragma unroll
    for (int nt = 0; nt < 15; ++nt) {
      v16h b = frag_BT(BshT, 16 * nt, 32, lane);
      acc[nt] = WMMA_F16(a, b, acc[nt]);
    }
    __syncthreads();
  }
#pragma unroll
  for (int nt = 0; nt < 15; ++nt) {
    int col = 16 * nt + (lane & 15);
    float bias = bf1[col];
#pragma unroll
    for (int v = 0; v < 8; ++v) {
      int row = rowBase + 16 * w + v + 8 * hl;
      H[(size_t)row * 256 + col] = (_Float16)fast_tanh(acc[nt][v] + bias);
    }
  }
  if (t < 128) {
    int row = rowBase + t;
#pragma unroll
    for (int c = 240; c < 256; ++c) H[(size_t)row * 256 + c] = (_Float16)0.0f;
  }
}

// =====================================================================
// Kernel 4: Hout = Hin + tanh(Hin @ Wf + bf); K padded to 256
// =====================================================================
__global__ __launch_bounds__(256) void k_fit_res(
    const _Float16* __restrict__ Hin, const _Float16* __restrict__ WT,
    const float* __restrict__ bf, _Float16* __restrict__ Hout)
{
  __shared__ _Float16 Ash[128 * 32];
  __shared__ _Float16 BshT[240 * 32];
  int t = threadIdx.x, lane = t & 31, w = t >> 5, hl = lane >> 4;
  int rowBase = blockIdx.x * 128;
  v8f zero = {};
  v8f acc[15];
#pragma unroll
  for (int i = 0; i < 15; ++i) acc[i] = zero;

  for (int kk = 0; kk < 8; ++kk) {
    {
      int row = t >> 1, seg = (t & 1) * 16;
      const _Float16* src = Hin + (size_t)(rowBase + row) * 256 + kk * 32 + seg;
      async_ld_b128(&Ash[row * 32 + seg],     src);
      async_ld_b128(&Ash[row * 32 + seg + 8], src + 8);
    }
    for (int c = t; c < 960; c += 256) {
      int row = c >> 2, seg = (c & 3) * 8;
      async_ld_b128(&BshT[row * 32 + seg], WT + (size_t)row * 256 + kk * 32 + seg);
    }
    wait_async0();
    __syncthreads();
    v16h a = frag_A(Ash, 16 * w, 32, lane);
#pragma unroll
    for (int nt = 0; nt < 15; ++nt) {
      v16h b = frag_BT(BshT, 16 * nt, 32, lane);
      acc[nt] = WMMA_F16(a, b, acc[nt]);
    }
    __syncthreads();
  }
#pragma unroll
  for (int nt = 0; nt < 15; ++nt) {
    int col = 16 * nt + (lane & 15);
    float bias = bf[col];
#pragma unroll
    for (int v = 0; v < 8; ++v) {
      int row = rowBase + 16 * w + v + 8 * hl;
      float hv = (float)Hin[(size_t)row * 256 + col];
      Hout[(size_t)row * 256 + col] = (_Float16)(hv + fast_tanh(acc[nt][v] + bias));
    }
  }
  if (t < 128) {
    int row = rowBase + t;
#pragma unroll
    for (int c = 240; c < 256; ++c) Hout[(size_t)row * 256 + c] = (_Float16)0.0f;
  }
}

// =====================================================================
// Kernel 5: pred = sum_n (H3[n] . Wfo + bfo + EBIAS)
// =====================================================================
__global__ __launch_bounds__(1024) void k_out(
    const _Float16* __restrict__ H, const float* __restrict__ Wfo,
    const float* __restrict__ bfo, float* __restrict__ out)
{
  __shared__ float red[1024];
  __shared__ float wsh[240];
  int t = threadIdx.x;
  if (t < 240) wsh[t] = Wfo[t];
  __syncthreads();
  float b = bfo[0];
  float local = 0.0f;
  for (int n = t; n < 4096; n += 1024) {
    const uint4* hp = (const uint4*)(H + (size_t)n * 256);
    float e = b;
#pragma unroll
    for (int cc = 0; cc < 30; ++cc) {
      uint4 pk = hp[cc];
      const _Float16* hv = (const _Float16*)&pk;
#pragma unroll
      for (int j = 0; j < 8; ++j) e += (float)hv[j] * wsh[cc * 8 + j];
    }
    local += e + (-93.5f);
  }
  red[t] = local;
  __syncthreads();
  for (int s = 512; s > 0; s >>= 1) {
    if (t < s) red[t] += red[t + s];
    __syncthreads();
  }
  if (t == 0) out[0] = red[0];
}

extern "C" void kernel_launch(void* const* d_in, const int* in_sizes, int n_in,
                              void* d_out, int out_size, void* d_ws, size_t ws_size,
                              hipStream_t stream) {
  (void)in_sizes; (void)n_in; (void)out_size; (void)ws_size;
  const float* coord = (const float*)d_in[0];
  const int*   nbrs  = (const int*)d_in[1];
  const float* We1 = (const float*)d_in[2];
  const float* be1 = (const float*)d_in[3];
  const float* We2 = (const float*)d_in[4];
  const float* be2 = (const float*)d_in[5];
  const float* We3 = (const float*)d_in[6];
  const float* be3 = (const float*)d_in[7];
  const float* Tbias = (const float*)d_in[8];
  const float* Wf1 = (const float*)d_in[9];
  const float* bf1 = (const float*)d_in[10];
  const float* Wf2 = (const float*)d_in[11];
  const float* bf2 = (const float*)d_in[12];
  const float* Wf3 = (const float*)d_in[13];
  const float* bf3 = (const float*)d_in[14];
  const float* Wfo = (const float*)d_in[15];
  const float* bfo = (const float*)d_in[16];

  char* ws = (char*)d_ws;
  float*    T  = (float*)ws;                             // 4096*512*4  = 8 MB (dead after k_descriptor)
  _Float16* D  = (_Float16*)(ws + (8u  << 20));          // 16 MB
  _Float16* H1 = (_Float16*)(ws + (24u << 20));          // 2 MB
  _Float16* H2 = (_Float16*)(ws + (26u << 20));          // 2 MB
  _Float16* H3 = (_Float16*)(ws + (28u << 20));          // 2 MB
  // prepped weights reuse the dead T region:
  _Float16* WT1 = (_Float16*)ws;                         // 240*2048*2 = 0.94 MB
  _Float16* WT2 = (_Float16*)(ws + (1u << 20));          // 240*256*2  = 120 KB
  _Float16* WT3 = (_Float16*)(ws + (1u << 20) + (256u << 10));

  k_embed<<<4096, 256, 0, stream>>>(coord, nbrs, We1, be1, We2, be2, We3, be3, T);
  k_descriptor<<<4096, 256, 0, stream>>>(T, Tbias, D);
  k_prep_w<<<256, 256, 0, stream>>>(Wf1, Wf2, Wf3, WT1, WT2, WT3);
  k_fit1<<<32, 256, 0, stream>>>(D, WT1, bf1, H1);
  k_fit_res<<<32, 256, 0, stream>>>(H1, WT2, bf2, H2);
  k_fit_res<<<32, 256, 0, stream>>>(H2, WT3, bf3, H3);
  k_out<<<1, 1024, 0, stream>>>(H3, Wfo, bfo, (float*)d_out);
}
